// HfMistral4MoE_17085379904041
// MI455X (gfx1250) — matrix operand order repaired and checked by reading, not compiled, not run on hardware
//
#include <hip/hip_runtime.h>
#include <math.h>
#include <stdint.h>

// ---------------- problem constants (from reference) ----------------
#define T_DIM 1024
#define H_DIM 2048
#define I_DIM 1408
#define E_NUM 8
#define NGRP  9          // 8 routed experts + 1 shared expert group

// ---------------- GEMM tiling ----------------
#define BM 256           // large M tile -> most experts need a single M tile
#define BN 64
#define BK 32
#define MT (T_DIM / BM)  // 4 M tiles per group
#define LDT 40           // padded LDS row stride in bf16 elements

typedef __attribute__((ext_vector_type(16))) __bf16 v16bf;
typedef __attribute__((ext_vector_type(2)))  __bf16 v2bf;
typedef __attribute__((ext_vector_type(8)))  float  v8f;

union Frag16 { v16bf v; unsigned int u[8]; };

// ---------------- gfx1250 async global->LDS copy (probe-guarded) ----------------
#if defined(__gfx1250__) && __has_builtin(__builtin_amdgcn_global_load_async_to_lds_b128)
#define HAVE_ASYNC_COPY 1
#endif

typedef int v4i_vs __attribute__((__vector_size__(16)));
typedef __attribute__((address_space(1))) v4i_vs* g_v4i_ptr;
typedef __attribute__((address_space(3))) v4i_vs* l_v4i_ptr;

__device__ __forceinline__ void copy16_g2l(const unsigned short* gsrc, unsigned short* ldst) {
#if defined(HAVE_ASYNC_COPY)
  __builtin_amdgcn_global_load_async_to_lds_b128(
      (g_v4i_ptr)(uintptr_t)gsrc,
      (l_v4i_ptr)(unsigned)(uintptr_t)ldst,
      0, 0);
#else
  *(uint4*)ldst = *(const uint4*)gsrc;
#endif
}

__device__ __forceinline__ void async_copy_wait() {
#if defined(HAVE_ASYNC_COPY)
#if __has_builtin(__builtin_amdgcn_s_wait_asynccnt)
  __builtin_amdgcn_s_wait_asynccnt(0);
#else
  asm volatile("s_wait_asynccnt 0" ::: "memory");
#endif
#endif
}

// ---------------- bf16 conversion via native converts ----------------
__device__ __forceinline__ unsigned short bf16_bits(float f) {
  __bf16 b = (__bf16)f;
  return __builtin_bit_cast(unsigned short, b);
}

__device__ __forceinline__ unsigned int pack2_bf16(float x, float y) {
  v2bf p;
  p[0] = (__bf16)x;
  p[1] = (__bf16)y;
  return __builtin_bit_cast(unsigned int, p);
}

__device__ __forceinline__ uint2 pack4_bf16(float4 v) {
  uint2 q;
  q.x = pack2_bf16(v.x, v.y);
  q.y = pack2_bf16(v.z, v.w);
  return q;
}

// A fragment: 16x32 bf16, M = lane%16.
// lane<16:  r<4 -> K=2r,2r+1 ; r>=4 -> K=2r+8,2r+9 ;  lane>=16: +8 on K.
__device__ __forceinline__ v16bf load_a_frag(const unsigned short* As, int m0, int lane) {
  Frag16 f;
  int m = m0 + (lane & 15);
  int kadd = (lane < 16) ? 0 : 8;
  const unsigned short* row = As + m * LDT;
#pragma unroll
  for (int r = 0; r < 8; ++r) {
    int k = 2 * r + ((r >= 4) ? 8 : 0) + kadd;
    f.u[r] = *(const unsigned int*)(row + k);
  }
  return f.v;
}

// B fragment: 32x16 bf16 staged as [N][K]. N = lane%16, same K order as A fragment (lanes>=16: +8 on K).
__device__ __forceinline__ v16bf load_b_frag(const unsigned short* Bs, int n0, int lane) {
  Frag16 f;
  int n = n0 + (lane & 15);
  int k0 = (lane < 16) ? 0 : 8;
  const unsigned short* row = Bs + n * LDT + k0;
#pragma unroll
  for (int r = 0; r < 8; ++r)
    f.u[r] = *(const unsigned int*)(row + 2 * r + ((r >= 4) ? 8 : 0));
  return f.v;
}

#define WMMA_BF16(A, B, C) \
  __builtin_amdgcn_wmma_f32_16x16x32_bf16(false, (A), false, (B), (short)0, (C), false, false)

// ---------------- routing ----------------
__global__ void init_cnt_kernel(int* cnt) {
  if (threadIdx.x < E_NUM) cnt[threadIdx.x] = 0;
}

__global__ void route_kernel(const float* __restrict__ x, const float* __restrict__ gw,
                             int* __restrict__ cnt, int* __restrict__ tok_list,
                             int* __restrict__ k_list, float* __restrict__ w_list) {
  int t = blockIdx.x * blockDim.x + threadIdx.x;
  if (t >= T_DIM) return;
  float acc[E_NUM];
#pragma unroll
  for (int e = 0; e < E_NUM; ++e) acc[e] = 0.f;
  const float4* xr = (const float4*)(x + (size_t)t * H_DIM);
  for (int h4 = 0; h4 < H_DIM / 4; ++h4) {
    float4 xv = xr[h4];
#pragma unroll
    for (int e = 0; e < E_NUM; ++e) {
      float4 g = ((const float4*)(gw + (size_t)e * H_DIM))[h4];
      acc[e] += xv.x * g.x + xv.y * g.y + xv.z * g.z + xv.w * g.w;
    }
  }
  int e0 = 0;
#pragma unroll
  for (int e = 1; e < E_NUM; ++e) if (acc[e] > acc[e0]) e0 = e;
  int e1 = (e0 == 0) ? 1 : 0;
#pragma unroll
  for (int e = 0; e < E_NUM; ++e)
    if (e != e0 && acc[e] > acc[e1]) e1 = e;
  float p1 = __expf(acc[e1] - acc[e0]);
  float w0 = __fdividef(1.f, 1.f + p1);
  float w1 = 1.f - w0;

  int s0 = atomicAdd(&cnt[e0], 1);
  tok_list[e0 * T_DIM + s0] = t; k_list[e0 * T_DIM + s0] = 0; w_list[e0 * T_DIM + s0] = w0;
  int s1 = atomicAdd(&cnt[e1], 1);
  tok_list[e1 * T_DIM + s1] = t; k_list[e1 * T_DIM + s1] = 1; w_list[e1 * T_DIM + s1] = w1;
}

// ---------------- fp32 -> bf16 activation cast ----------------
__global__ void cast_bf16_kernel(const float* __restrict__ x, unsigned short* __restrict__ xb) {
  int i = blockIdx.x * blockDim.x + threadIdx.x;
  if (i >= (T_DIM * H_DIM) / 4) return;
  ((uint2*)xb)[i] = pack4_bf16(((const float4*)x)[i]);
}

// ---------------- grouped GEMM 1: gate & up, silu(g)*u -> hb (bf16) ----------------
__global__ __launch_bounds__(256) void mlp1_kernel(
    const unsigned short* __restrict__ xb,
    const float* __restrict__ eg, const float* __restrict__ eu,
    const float* __restrict__ sg, const float* __restrict__ su,
    const int* __restrict__ cnt, const int* __restrict__ tok_list,
    unsigned short* __restrict__ hb) {
  int gy = blockIdx.y;
  int g = gy >> 2;              // group 0..8
  int tm = gy & (MT - 1);       // M tile within group
  int rows = (g < E_NUM) ? cnt[g] : T_DIM;
  if (tm * BM >= rows) return;

  const float* Wg = (g < E_NUM) ? eg + (size_t)g * I_DIM * H_DIM : sg;
  const float* Wu = (g < E_NUM) ? eu + (size_t)g * I_DIM * H_DIM : su;
  int n0g = blockIdx.x * BN;

  __shared__ unsigned short As[2][BM * LDT];
  __shared__ unsigned short Bg[2][BN * LDT];
  __shared__ unsigned short Bu[2][BN * LDT];

  int tid = threadIdx.x;
  int lane = tid & 31;
  int wave = tid >> 5;
  int wm = (wave & 3) * 64;     // 4 waves along M, 64 rows each
  int wn = (wave >> 2) * 32;    // 2 waves along N, 32 cols each

  // ---- hoisted per-thread staging addresses (loop-invariant) ----
  const unsigned short* asrc[4];
  int adoff[4];
#pragma unroll
  for (int c = 0; c < 4; ++c) {
    int m = (tid >> 2) + 64 * c;
    int kk = (tid & 3) * 8;
    int slot = tm * BM + m;
    int tok = (g == E_NUM) ? slot : (slot < rows ? tok_list[g * T_DIM + slot] : 0);
    asrc[c] = xb + (size_t)tok * H_DIM + kk;   // pad rows gather token 0 (never combined)
    adoff[c] = m * LDT + kk;
  }
  const float *bgsrc[2], *busrc[2];
  int bdoff[2];
#pragma unroll
  for (int c = 0; c < 2; ++c) {
    int idx = (tid + c * 256) * 4;
    int n = idx >> 5, kk = idx & 31;
    bgsrc[c] = Wg + (size_t)(n0g + n) * H_DIM + kk;
    busrc[c] = Wu + (size_t)(n0g + n) * H_DIM + kk;
    bdoff[c] = n * LDT + kk;
  }

  v8f zero = {0.f, 0.f, 0.f, 0.f, 0.f, 0.f, 0.f, 0.f};
  v8f ag[4][2], au[4][2];
#pragma unroll
  for (int i = 0; i < 4; ++i)
#pragma unroll
    for (int j = 0; j < 2; ++j) { ag[i][j] = zero; au[i][j] = zero; }

  // ---- prologue: stage K tile 0 into buffer 0 ----
#pragma unroll
  for (int c = 0; c < 4; ++c) copy16_g2l(asrc[c], &As[0][adoff[c]]);
#pragma unroll
  for (int c = 0; c < 2; ++c) {
    *(uint2*)&Bg[0][bdoff[c]] = pack4_bf16(*(const float4*)bgsrc[c]);
    *(uint2*)&Bu[0][bdoff[c]] = pack4_bf16(*(const float4*)busrc[c]);
  }
  async_copy_wait();
  __syncthreads();

  const int NK = H_DIM / BK;
  for (int kt = 0; kt < NK; ++kt) {
    int cur = kt & 1, nxt = cur ^ 1;
    int k1 = (kt + 1) * BK;
    bool hn = (kt + 1 < NK);          // uniform

    if (hn) {
      // async-stage next A tile into the other buffer (overlaps with WMMA below)
#pragma unroll
      for (int c = 0; c < 4; ++c) copy16_g2l(asrc[c] + k1, &As[nxt][adoff[c]]);
    }
    float4 rg[2], ru[2];
    if (hn) {
      // issue next B weight loads now; consumed after the WMMAs
#pragma unroll
      for (int c = 0; c < 2; ++c) {
        rg[c] = *(const float4*)(bgsrc[c] + k1);
        ru[c] = *(const float4*)(busrc[c] + k1);
        if (k1 + BK < H_DIM) {
          __builtin_prefetch(bgsrc[c] + k1 + BK, 0, 1);
          __builtin_prefetch(busrc[c] + k1 + BK, 0, 1);
        }
      }
    }

    // ---- compute on current buffer ----
    v16bf bg0 = load_b_frag(Bg[cur], wn, lane);
    v16bf bg1 = load_b_frag(Bg[cur], wn + 16, lane);
    v16bf bu0 = load_b_frag(Bu[cur], wn, lane);
    v16bf bu1 = load_b_frag(Bu[cur], wn + 16, lane);
#pragma unroll
    for (int i = 0; i < 4; ++i) {
      v16bf a = load_a_frag(As[cur], wm + i * 16, lane);
      ag[i][0] = WMMA_BF16(a, bg0, ag[i][0]);
      ag[i][1] = WMMA_BF16(a, bg1, ag[i][1]);
      au[i][0] = WMMA_BF16(a, bu0, au[i][0]);
      au[i][1] = WMMA_BF16(a, bu1, au[i][1]);
    }

    if (hn) {
      // convert + store next B tile into the other buffer
#pragma unroll
      for (int c = 0; c < 2; ++c) {
        *(uint2*)&Bg[nxt][bdoff[c]] = pack4_bf16(rg[c]);
        *(uint2*)&Bu[nxt][bdoff[c]] = pack4_bf16(ru[c]);
      }
    }
    async_copy_wait();
    __syncthreads();
  }

  // epilogue: silu(g)*u -> bf16 scratch
  int half = lane >> 4;
  int nn = lane & 15;
#pragma unroll
  for (int i = 0; i < 4; ++i)
#pragma unroll
    for (int j = 0; j < 2; ++j) {
      int n = n0g + wn + j * 16 + nn;
#pragma unroll
      for (int r = 0; r < 8; ++r) {
        int m = tm * BM + wm + i * 16 + half * 8 + r;
        float gv = ag[i][j][r];
        float uv = au[i][j][r];
        float hv = __fdividef(gv, 1.f + __expf(-gv)) * uv;
        hb[((size_t)g * T_DIM + m) * I_DIM + n] = bf16_bits(hv);
      }
    }
}

// ---------------- grouped GEMM 2: down projection + weighted scatter ----------------
__global__ __launch_bounds__(256) void mlp2_kernel(
    const unsigned short* __restrict__ hb,
    const float* __restrict__ ed, const float* __restrict__ sd,
    const int* __restrict__ cnt, const int* __restrict__ tok_list,
    const int* __restrict__ k_list, const float* __restrict__ w_list,
    float* __restrict__ Y2, float* __restrict__ out) {
  int gy = blockIdx.y;
  int g = gy >> 2;
  int tm = gy & (MT - 1);
  int rows = (g < E_NUM) ? cnt[g] : T_DIM;
  if (tm * BM >= rows) return;

  const float* Wd = (g < E_NUM) ? ed + (size_t)g * H_DIM * I_DIM : sd;
  int n0g = blockIdx.x * BN;

  __shared__ unsigned short As[2][BM * LDT];
  __shared__ unsigned short Bd[2][BN * LDT];
  __shared__ int rtok[BM];
  __shared__ int rk[BM];
  __shared__ float rw[BM];

  int tid = threadIdx.x;
  {
    // scatter metadata for epilogue (visible after the loop's barriers)
    int slot = tm * BM + tid;
    int tok = -1, kk = 0; float w = 0.f;
    if (g < E_NUM && slot < rows) {
      tok = tok_list[g * T_DIM + slot];
      kk  = k_list[g * T_DIM + slot];
      w   = w_list[g * T_DIM + slot];
    }
    rtok[tid] = tok; rk[tid] = kk; rw[tid] = w;
  }

  int lane = tid & 31;
  int wave = tid >> 5;
  int wm = (wave & 3) * 64;
  int wn = (wave >> 2) * 32;

  // ---- hoisted per-thread staging addresses ----
  const unsigned short* asrc[4];
  int adoff[4];
#pragma unroll
  for (int c = 0; c < 4; ++c) {
    int m = (tid >> 2) + 64 * c;
    int kk = (tid & 3) * 8;
    asrc[c] = hb + ((size_t)g * T_DIM + tm * BM + m) * I_DIM + kk;
    adoff[c] = m * LDT + kk;
  }
  const float* bdsrc[2];
  int bdoff[2];
#pragma unroll
  for (int c = 0; c < 2; ++c) {
    int idx = (tid + c * 256) * 4;
    int n = idx >> 5, kk = idx & 31;
    bdsrc[c] = Wd + (size_t)(n0g + n) * I_DIM + kk;
    bdoff[c] = n * LDT + kk;
  }

  v8f zero = {0.f, 0.f, 0.f, 0.f, 0.f, 0.f, 0.f, 0.f};
  v8f acc[4][2];
#pragma unroll
  for (int i = 0; i < 4; ++i)
#pragma unroll
    for (int j = 0; j < 2; ++j) acc[i][j] = zero;

  // ---- prologue ----
#pragma unroll
  for (int c = 0; c < 4; ++c) copy16_g2l(asrc[c], &As[0][adoff[c]]);
#pragma unroll
  for (int c = 0; c < 2; ++c)
    *(uint2*)&Bd[0][bdoff[c]] = pack4_bf16(*(const float4*)bdsrc[c]);
  async_copy_wait();
  __syncthreads();

  const int NK = I_DIM / BK;
  for (int kt = 0; kt < NK; ++kt) {
    int cur = kt & 1, nxt = cur ^ 1;
    int k1 = (kt + 1) * BK;
    bool hn = (kt + 1 < NK);

    if (hn) {
#pragma unroll
      for (int c = 0; c < 4; ++c) copy16_g2l(asrc[c] + k1, &As[nxt][adoff[c]]);
    }
    float4 rd[2];
    if (hn) {
#pragma unroll
      for (int c = 0; c < 2; ++c) {
        rd[c] = *(const float4*)(bdsrc[c] + k1);
        if (k1 + BK < I_DIM)
          __builtin_prefetch(bdsrc[c] + k1 + BK, 0, 1);
      }
    }

    v16bf b0 = load_b_frag(Bd[cur], wn, lane);
    v16bf b1 = load_b_frag(Bd[cur], wn + 16, lane);
#pragma unroll
    for (int i = 0; i < 4; ++i) {
      v16bf a = load_a_frag(As[cur], wm + i * 16, lane);
      acc[i][0] = WMMA_BF16(a, b0, acc[i][0]);
      acc[i][1] = WMMA_BF16(a, b1, acc[i][1]);
    }

    if (hn) {
#pragma unroll
      for (int c = 0; c < 2; ++c)
        *(uint2*)&Bd[nxt][bdoff[c]] = pack4_bf16(rd[c]);
    }
    async_copy_wait();
    __syncthreads();
  }

  int half = lane >> 4;
  int nn = lane & 15;
#pragma unroll
  for (int i = 0; i < 4; ++i)
#pragma unroll
    for (int j = 0; j < 2; ++j) {
      int n = n0g + wn + j * 16 + nn;
#pragma unroll
      for (int r = 0; r < 8; ++r) {
        int mloc = wm + i * 16 + half * 8 + r;
        float val = acc[i][j][r];
        if (g == E_NUM) {
          out[(size_t)(tm * BM + mloc) * H_DIM + n] = val;  // shared expert -> out
        } else {
          int tok = rtok[mloc];
          if (tok >= 0)
            Y2[((size_t)tok * 2 + rk[mloc]) * H_DIM + n] = rw[mloc] * val;
        }
      }
    }
}

// ---------------- final combine: out += w0*y_e0 + w1*y_e1 (deterministic) ----------------
__global__ void combine_kernel(const float* __restrict__ Y2, float* __restrict__ out) {
  int i = blockIdx.x * blockDim.x + threadIdx.x;
  const int n4 = (T_DIM * H_DIM) / 4;
  if (i >= n4) return;
  const int h4pr = H_DIM / 4;
  int t = i / h4pr, h4 = i % h4pr;
  const float4* y0 = (const float4*)(Y2 + (size_t)(t * 2) * H_DIM);
  const float4* y1 = (const float4*)(Y2 + (size_t)(t * 2 + 1) * H_DIM);
  float4 a = y0[h4], b = y1[h4], c = ((const float4*)out)[i];
  float4 o;
  o.x = a.x + b.x + c.x;
  o.y = a.y + b.y + c.y;
  o.z = a.z + b.z + c.z;
  o.w = a.w + b.w + c.w;
  ((float4*)out)[i] = o;
}

// ---------------- launcher ----------------
extern "C" void kernel_launch(void* const* d_in, const int* in_sizes, int n_in,
                              void* d_out, int out_size, void* d_ws, size_t ws_size,
                              hipStream_t stream) {
  (void)in_sizes; (void)n_in; (void)out_size; (void)ws_size;

  const float* x  = (const float*)d_in[0];   // [T, H]
  const float* gw = (const float*)d_in[1];   // [E, H]
  const float* eg = (const float*)d_in[2];   // [E, I, H]
  const float* eu = (const float*)d_in[3];   // [E, I, H]
  const float* ed = (const float*)d_in[4];   // [E, H, I]
  const float* sg = (const float*)d_in[5];   // [I, H]
  const float* su = (const float*)d_in[6];   // [I, H]
  const float* sd = (const float*)d_in[7];   // [H, I]
  float* out = (float*)d_out;

  char* ws = (char*)d_ws;
  size_t off = 0;
  auto alloc = [&](size_t bytes) -> void* {
    off = (off + 255) & ~(size_t)255;
    void* p = ws + off;
    off += bytes;
    return p;
  };
  unsigned short* xb = (unsigned short*)alloc((size_t)T_DIM * H_DIM * 2);         // 4 MB
  unsigned short* hb = (unsigned short*)alloc((size_t)NGRP * T_DIM * I_DIM * 2);  // 26 MB
  float* Y2 = (float*)alloc((size_t)T_DIM * 2 * H_DIM * 4);                       // 17 MB
  int*   cnt      = (int*)alloc(E_NUM * 4);
  int*   tok_list = (int*)alloc((size_t)E_NUM * T_DIM * 4);
  int*   k_list   = (int*)alloc((size_t)E_NUM * T_DIM * 4);
  float* w_list   = (float*)alloc((size_t)E_NUM * T_DIM * 4);

  init_cnt_kernel<<<1, 32, 0, stream>>>(cnt);
  route_kernel<<<T_DIM / 256, 256, 0, stream>>>(x, gw, cnt, tok_list, k_list, w_list);
  cast_bf16_kernel<<<(T_DIM * H_DIM / 4 + 255) / 256, 256, 0, stream>>>(x, xb);
  mlp1_kernel<<<dim3(I_DIM / BN, NGRP * MT), 256, 0, stream>>>(xb, eg, eu, sg, su, cnt,
                                                               tok_list, hb);
  mlp2_kernel<<<dim3(H_DIM / BN, NGRP * MT), 256, 0, stream>>>(hb, ed, sd, cnt, tok_list,
                                                               k_list, w_list, Y2, out);
  combine_kernel<<<(T_DIM * H_DIM / 4 + 255) / 256, 256, 0, stream>>>(Y2, out);
}
